// Qwen3MOEMLP_403726926300
// MI455X (gfx1250) — matrix operand order repaired and checked by reading, not compiled, not run on hardware
//
#include <hip/hip_runtime.h>
#include <hip/hip_bf16.h>

#define N_EMBED  1024
#define I_SIZE   1408
#define N_EXP    8
#define NTOK     8192      // B*T = 4*2048
#define NSLOTS   (NTOK*2)  // top-2
#define BM       64
#define BN       64
#define BK       32
#define LDP      (BK + 8)  // padded LDS row stride (80B, keeps 16B chunks aligned)

typedef __bf16 bf16_t;
typedef __attribute__((ext_vector_type(16))) __bf16 v16bf;
typedef __attribute__((ext_vector_type(8)))  __bf16 v8bf;
typedef __attribute__((ext_vector_type(8)))  float  v8f;

union ABfrag { v16bf v; v8bf h[2]; };

// A-fragment (16x32 bf16, MxK): lanes 0-15 rows M=0..15 with K=[0..8)+[16..24),
// lanes 16-31 same rows with K=[8..16)+[24..32). Tile is K-major, stride ldt.
__device__ __forceinline__ v16bf frag_a(const bf16_t* tile, int ldt, int row0, int lane) {
  const bf16_t* rp = tile + (row0 + (lane & 15)) * ldt + ((lane & 16) ? 8 : 0);
  ABfrag f;
  f.h[0] = *(const v8bf*)(rp);
  f.h[1] = *(const v8bf*)(rp + 16);
  return f.v;
}

// B-fragment (32x16 bf16, KxN) from an N-major (row = output col, K contiguous)
// tile: lanes 0-15 col N with K=[0..8)+[16..24), lanes 16-31 col N with K=[8..16)+[24..32).
__device__ __forceinline__ v16bf frag_b(const bf16_t* tile, int ldt, int col0, int lane) {
  const bf16_t* rp = tile + (col0 + (lane & 15)) * ldt + ((lane & 16) ? 8 : 0);
  ABfrag f;
  f.h[0] = *(const v8bf*)(rp);
  f.h[1] = *(const v8bf*)(rp + 16);
  return f.v;
}

__device__ __forceinline__ v8f vzero8() { v8f z = {0.f,0.f,0.f,0.f,0.f,0.f,0.f,0.f}; return z; }

// fast silu factor: sigmoid(v) via v_exp_f32 + v_rcp_f32 (no IEEE divide chain)
__device__ __forceinline__ float fast_sigmoid(float v) {
  return __builtin_amdgcn_rcpf(1.f + __expf(-v));
}

// ---------------- kernel 0: zero output + counters ----------------
__global__ __launch_bounds__(256) void moe_init(float* __restrict__ out, int* __restrict__ counts) {
  int i = blockIdx.x * blockDim.x + threadIdx.x;
  if (i < N_EXP) counts[i] = 0;
  const int total = NTOK * N_EMBED;
  for (int j = i; j < total; j += gridDim.x * blockDim.x) out[j] = 0.f;
}

// ---------------- kernel 1: router (one wave32 per token) ----------------
__global__ __launch_bounds__(256)
void moe_router(const float* __restrict__ x, const float* __restrict__ gate_w,
                int2* __restrict__ epair, float2* __restrict__ wpair, int* __restrict__ counts) {
  const int tok  = (blockIdx.x * blockDim.x + threadIdx.x) >> 5;
  const int lane = threadIdx.x & 31;
  if (tok >= NTOK) return;
  const float* xr = x + (size_t)tok * N_EMBED;
  float s[N_EXP];
  #pragma unroll
  for (int e = 0; e < N_EXP; ++e) s[e] = 0.f;
  for (int d = lane; d < N_EMBED; d += 32) {
    float xv = xr[d];
    #pragma unroll
    for (int e = 0; e < N_EXP; ++e) s[e] += xv * gate_w[e * N_EMBED + d];
  }
  #pragma unroll
  for (int e = 0; e < N_EXP; ++e)
    #pragma unroll
    for (int off = 16; off > 0; off >>= 1)
      s[e] += __shfl_xor(s[e], off, 32);
  if (lane == 0) {
    int i0 = 0; float v0 = s[0];
    #pragma unroll
    for (int e = 1; e < N_EXP; ++e) if (s[e] > v0) { v0 = s[e]; i0 = e; }
    int i1 = (i0 == 0) ? 1 : 0; float v1 = s[i1];
    #pragma unroll
    for (int e = 0; e < N_EXP; ++e) if (e != i0 && s[e] > v1) { v1 = s[e]; i1 = e; }
    float p0 = 1.f / (1.f + __expf(v1 - v0));   // softmax over the two selected scores
    epair[tok] = make_int2(i0, i1);
    wpair[tok] = make_float2(p0, 1.f - p0);
    atomicAdd(&counts[i0], 1);
    atomicAdd(&counts[i1], 1);
  }
}

// ---------------- kernel 2: tiny exclusive scan ----------------
__global__ void moe_scan(const int* __restrict__ counts, int* __restrict__ offsets,
                         int* __restrict__ cursor) {
  if (threadIdx.x == 0 && blockIdx.x == 0) {
    int acc = 0;
    for (int e = 0; e < N_EXP; ++e) { offsets[e] = acc; cursor[e] = acc; acc += counts[e]; }
    offsets[N_EXP] = acc;
  }
}

// ---------------- kernel 3: scatter tokens into per-expert slot lists ----------------
__global__ __launch_bounds__(256)
void moe_scatter(const int2* __restrict__ epair, const float2* __restrict__ wpair,
                 int* __restrict__ cursor, int* __restrict__ token_map,
                 float* __restrict__ weight_map) {
  int t = blockIdx.x * blockDim.x + threadIdx.x;
  if (t >= NTOK) return;
  int2 e2 = epair[t]; float2 w2 = wpair[t];
  int s0 = atomicAdd(&cursor[e2.x], 1);
  token_map[s0] = t; weight_map[s0] = w2.x;
  int s1 = atomicAdd(&cursor[e2.y], 1);
  token_map[s1] = t; weight_map[s1] = w2.y;
}

// ---------------- kernel 4: GEMM1 + fused SiLU-gate ----------------
// h[slot, 0:I] = (x_g @ w_in[e,0:I,:]^T) * silu(x_g @ w_in[e,I:2I,:]^T), bf16 out.
// k-loop is a register-prefetch pipeline: the next tile's global loads are issued
// right after the LDS-publish barrier so HBM latency overlaps the 8 WMMAs.
__global__ __launch_bounds__(256)
void moe_gemm1_silu(const float* __restrict__ x, const float* __restrict__ w_in,
                    const int* __restrict__ token_map, const int* __restrict__ offsets,
                    bf16_t* __restrict__ H) {
  const int e     = blockIdx.z;
  const int mtile = blockIdx.x;
  const int ntile = blockIdx.y;                 // 0..21 over I
  const int base  = offsets[e];
  const int cnt   = offsets[e + 1] - base;
  if (mtile * BM >= cnt) return;

  __shared__ bf16_t As[BM][LDP];
  __shared__ bf16_t Bu[BN][LDP];
  __shared__ bf16_t Bv[BN][LDP];

  const int tid  = threadIdx.x;
  const int lane = tid & 31;
  const int wave = tid >> 5;
  const int wm   = (wave >> 1) * 32;
  const int wn   = (wave & 1) * 32;

  // staging assignment: 256 threads = 64 rows x 4 chunks of 8 elements
  const int srow = tid >> 2;
  const int scol = (tid & 3) * 8;
  const int maxslot = base + cnt - 1;
  int slot = base + mtile * BM + srow;
  const int tok = token_map[slot <= maxslot ? slot : maxslot];  // clamp partial tile
  const float* arow  = x + (size_t)tok * N_EMBED + scol;
  const float* burow = w_in + ((size_t)e * 2 * I_SIZE + (size_t)(ntile * BN + srow)) * N_EMBED + scol;
  const float* bvrow = burow + (size_t)I_SIZE * N_EMBED;

  v8f acc_u[2][2], acc_v[2][2];
  #pragma unroll
  for (int sm = 0; sm < 2; ++sm)
    #pragma unroll
    for (int sn = 0; sn < 2; ++sn) { acc_u[sm][sn] = vzero8(); acc_v[sm][sn] = vzero8(); }

  v8bf ta, tu, tv;
  auto prefetch = [&](int k0) {
    #pragma unroll
    for (int j = 0; j < 8; ++j) {
      ta[j] = (bf16_t)arow[k0 + j];
      tu[j] = (bf16_t)burow[k0 + j];
      tv[j] = (bf16_t)bvrow[k0 + j];
    }
  };
  prefetch(0);

  for (int k0 = 0; k0 < N_EMBED; k0 += BK) {
    __syncthreads();                      // prior iteration's ds_loads done
    *(v8bf*)&As[srow][scol] = ta;
    *(v8bf*)&Bu[srow][scol] = tu;
    *(v8bf*)&Bv[srow][scol] = tv;
    __syncthreads();                      // tiles published
    if (k0 + BK < N_EMBED) prefetch(k0 + BK);   // issue next global loads early

    #pragma unroll
    for (int sm = 0; sm < 2; ++sm) {
      v16bf a = frag_a(&As[0][0], LDP, wm + sm * 16, lane);
      #pragma unroll
      for (int sn = 0; sn < 2; ++sn) {
        v16bf fu = frag_b(&Bu[0][0], LDP, wn + sn * 16, lane);
        v16bf fv = frag_b(&Bv[0][0], LDP, wn + sn * 16, lane);
        acc_u[sm][sn] = __builtin_amdgcn_wmma_f32_16x16x32_bf16(
            false, a, false, fu, (short)0, acc_u[sm][sn], false, false);
        acc_v[sm][sn] = __builtin_amdgcn_wmma_f32_16x16x32_bf16(
            false, a, false, fv, (short)0, acc_v[sm][sn], false, false);
      }
    }
  }

  // epilogue: C/D layout -> row = wm + sm*16 + (lane>=16 ? 8:0) + i, col = wn + sn*16 + (lane&15)
  const int rb = mtile * BM + wm + ((lane & 16) ? 8 : 0);
  const int cb = ntile * BN + wn + (lane & 15);
  #pragma unroll
  for (int sm = 0; sm < 2; ++sm)
    #pragma unroll
    for (int sn = 0; sn < 2; ++sn)
      #pragma unroll
      for (int i = 0; i < 8; ++i) {
        int r = rb + sm * 16 + i;
        if (r < cnt) {
          float u = acc_u[sm][sn][i];
          float v = acc_v[sm][sn][i];
          float h = u * v * fast_sigmoid(v);          // u * silu(v)
          H[(size_t)(base + r) * I_SIZE + cb + sn * 16] = (bf16_t)h;
        }
      }
}

// ---------------- kernel 5: GEMM2 + weighted atomic scatter ----------------
// out[token] += weight * (h[slot] @ w_out[e]^T)
__global__ __launch_bounds__(256)
void moe_gemm2_scatter(const bf16_t* __restrict__ H, const float* __restrict__ w_out,
                       const int* __restrict__ token_map, const float* __restrict__ weight_map,
                       const int* __restrict__ offsets, float* __restrict__ out) {
  const int e     = blockIdx.z;
  const int mtile = blockIdx.x;
  const int ntile = blockIdx.y;                 // 0..15 over D
  const int base  = offsets[e];
  const int cnt   = offsets[e + 1] - base;
  if (mtile * BM >= cnt) return;

  __shared__ bf16_t As[BM][LDP];
  __shared__ bf16_t Bs[BN][LDP];

  const int tid  = threadIdx.x;
  const int lane = tid & 31;
  const int wave = tid >> 5;
  const int wm   = (wave >> 1) * 32;
  const int wn   = (wave & 1) * 32;

  const int srow = tid >> 2;
  const int scol = (tid & 3) * 8;
  const int maxslot = base + cnt - 1;
  int slot = base + mtile * BM + srow;
  if (slot > maxslot) slot = maxslot;
  const bf16_t* arow = H + (size_t)slot * I_SIZE + scol;
  const float*  brow = w_out + ((size_t)e * N_EMBED + (size_t)(ntile * BN + srow)) * I_SIZE + scol;

  v8f acc[2][2];
  #pragma unroll
  for (int sm = 0; sm < 2; ++sm)
    #pragma unroll
    for (int sn = 0; sn < 2; ++sn) acc[sm][sn] = vzero8();

  v8bf ta, tb;
  auto prefetch = [&](int k0) {
    ta = *(const v8bf*)(arow + k0);               // H is already bf16
    #pragma unroll
    for (int j = 0; j < 8; ++j) tb[j] = (bf16_t)brow[k0 + j];
  };
  prefetch(0);

  for (int k0 = 0; k0 < I_SIZE; k0 += BK) {
    __syncthreads();
    *(v8bf*)&As[srow][scol] = ta;
    *(v8bf*)&Bs[srow][scol] = tb;
    __syncthreads();
    if (k0 + BK < I_SIZE) prefetch(k0 + BK);

    #pragma unroll
    for (int sm = 0; sm < 2; ++sm) {
      v16bf a = frag_a(&As[0][0], LDP, wm + sm * 16, lane);
      #pragma unroll
      for (int sn = 0; sn < 2; ++sn) {
        v16bf b = frag_b(&Bs[0][0], LDP, wn + sn * 16, lane);
        acc[sm][sn] = __builtin_amdgcn_wmma_f32_16x16x32_bf16(
            false, a, false, b, (short)0, acc[sm][sn], false, false);
      }
    }
  }

  const int rb = mtile * BM + wm + ((lane & 16) ? 8 : 0);
  const int cb = ntile * BN + wn + (lane & 15);
  #pragma unroll
  for (int sm = 0; sm < 2; ++sm)
    #pragma unroll
    for (int i = 0; i < 8; ++i) {
      int r = rb + sm * 16 + i;
      if (r < cnt) {
        int   s  = base + r;
        int   tk = token_map[s];
        float w  = weight_map[s];
        float* orow = out + (size_t)tk * N_EMBED;
        #pragma unroll
        for (int sn = 0; sn < 2; ++sn)
          atomicAdd(&orow[cb + sn * 16], w * acc[sm][sn][i]);
      }
    }
}

// ---------------- host-side launch ----------------
extern "C" void kernel_launch(void* const* d_in, const int* in_sizes, int n_in,
                              void* d_out, int out_size, void* d_ws, size_t ws_size,
                              hipStream_t stream) {
  (void)in_sizes; (void)n_in; (void)out_size; (void)ws_size;
  const float* x      = (const float*)d_in[0];
  const float* gate_w = (const float*)d_in[1];
  const float* w_in   = (const float*)d_in[2];
  const float* w_out  = (const float*)d_in[3];
  float* out = (float*)d_out;

  char* ws = (char*)d_ws;
  int*    counts     = (int*)(ws + 0);
  int*    offsets    = (int*)(ws + 64);
  int*    cursor     = (int*)(ws + 128);
  int2*   epair      = (int2*)(ws + 256);
  float2* wpair      = (float2*)(ws + 256 + 1 * 65536);
  int*    token_map  = (int*)(ws + 256 + 2 * 65536);
  float*  weight_map = (float*)(ws + 256 + 3 * 65536);
  bf16_t* H          = (bf16_t*)(ws + 256 + 4 * 65536);  // NSLOTS * I_SIZE bf16 (~46 MB)

  moe_init<<<1024, 256, 0, stream>>>(out, counts);
  moe_router<<<NTOK / 8, 256, 0, stream>>>(x, gate_w, epair, wpair, counts);
  moe_scan<<<1, 32, 0, stream>>>(counts, offsets, cursor);
  moe_scatter<<<NTOK / 256, 256, 0, stream>>>(epair, wpair, cursor, token_map, weight_map);
  moe_gemm1_silu<<<dim3(NTOK / BM, I_SIZE / BN, N_EXP), 256, 0, stream>>>(
      x, w_in, token_map, offsets, H);
  moe_gemm2_scatter<<<dim3(NTOK / BM, N_EMBED / BN, N_EXP), 256, 0, stream>>>(
      H, w_out, token_map, weight_map, offsets, out);
}
